// OreoMemoryMLP_7112465842282
// MI455X (gfx1250) — compile-verified, matrix operand order checked by reading, not run on hardware
//
#include <hip/hip_runtime.h>
#include <math.h>

// ---------------------------------------------------------------------------
// Fused MLP + top-2 memory attention for MI455X (gfx1250), bf16 WMMA pipeline.
// x-tile staged once in LDS (bf16, NT global loads); weights pre-packed into
// WMMA B-fragment layout (L2-resident); all intermediates live in LDS.
// ---------------------------------------------------------------------------

typedef __attribute__((ext_vector_type(4)))  float   v4f;
typedef __attribute__((ext_vector_type(8)))  float   v8f;
typedef __attribute__((ext_vector_type(4)))  __bf16  v4bf;
typedef __attribute__((ext_vector_type(8)))  __bf16  v8bf;
typedef __attribute__((ext_vector_type(16))) __bf16  v16bf;

#define TAU_F    0.7f
#define BATCH    131072
#define DIN      512
#define H1       384
#define LAT      128
#define NSLOT    32
#define MBLK     128     // rows per block
#define XS_LD    520     // 512 + 8 bf16 pad (bank-conflict avoidance)
#define Z1_LD    392     // 384 + 8 bf16 pad
#define ZC_LD    264     // 256 + 8 bf16 pad
#define LG_LD    36      // 32 + 4 f32 pad

// LDS map (bytes):  [0 .. 133120)  xs  [128][520] bf16   (reused after L1:
//                                      zcs [128][264] bf16 @0, lgs [128][36] f32 @67584)
//                   [133120 .. 233472) z1s [128][392] bf16
#define XS_BYTES  (MBLK * XS_LD * 2)
#define LDS_TOTAL (XS_BYTES + MBLK * Z1_LD * 2)

// ---- scalar bf16 conversions (RNE) ----------------------------------------
__device__ __forceinline__ __bf16 f2bf(float f) {
  unsigned u = __float_as_uint(f);
  u = u + 0x7FFFu + ((u >> 16) & 1u);
  unsigned short s = (unsigned short)(u >> 16);
  __bf16 b; __builtin_memcpy(&b, &s, 2); return b;
}
__device__ __forceinline__ float bf2f(__bf16 b) {
  unsigned short s; __builtin_memcpy(&s, &b, 2);
  return __uint_as_float(((unsigned)s) << 16);
}
__device__ __forceinline__ float gelu_erf(float t) {
  return 0.5f * t * (1.0f + erff(t * 0.70710678118654752f));
}
__device__ __forceinline__ v8f wmma_bf16(v16bf a, v16bf b, v8f c) {
  return __builtin_amdgcn_wmma_f32_16x16x32_bf16(
      /*neg_a=*/false, a, /*neg_b=*/false, b,
      /*c_mod=*/(short)0, c, /*reuse_a=*/false, /*reuse_b=*/false);
}
__device__ __forceinline__ v16bf cat16(v8bf lo, v8bf hi) {
  v16bf r;
#pragma unroll
  for (int i = 0; i < 8; ++i) { r[i] = lo[i]; r[i + 8] = hi[i]; }
  return r;
}

// A-fragment (16x32 bf16, M x K) from row-major bf16 LDS tile.
// ISA layout: lane holds row (lane&15); lanes>=16 take K offset +8;
// VGPR0..3 = K[koff..koff+7], VGPR4..7 = K[koff+16..koff+23].
__device__ __forceinline__ v16bf load_a_lds(const __bf16* base, int ld, int kt, int lane) {
  const __bf16* p = base + (size_t)(lane & 15) * ld + kt * 32 + ((lane < 16) ? 0 : 8);
  return cat16(*(const v8bf*)p, *(const v8bf*)(p + 16));
}
// B-fragment from the pre-packed fragment buffer: lane-contiguous 16 bf16.
__device__ __forceinline__ v16bf load_b_frag(const __bf16* frags, int tile, int lane) {
  const __bf16* p = frags + ((size_t)tile * 32 + lane) * 16;
  return cat16(*(const v8bf*)p, *(const v8bf*)(p + 8));
}

// ---------------------------------------------------------------------------
// Prepass: pack f32 weight [K x N] (or [N x K] if transpose) into WMMA
// B-fragment layout: out[((kt*(N/16)+nt)*32 + lane)*16 + e], e = 2v+half.
// ---------------------------------------------------------------------------
__global__ void pack_b_frags(const float* __restrict__ W, __bf16* __restrict__ out,
                             int Kdim, int N, float scale, int transpose, int total) {
  int id = blockIdx.x * blockDim.x + threadIdx.x;
  if (id >= total) return;
  int tile = id >> 9;       // 512 elements per 32x16 tile
  int rem  = id & 511;
  int lane = rem >> 4;
  int e    = rem & 15;
  int ntn  = N >> 4;
  int kt = tile / ntn;
  int nt = tile - kt * ntn;
  int v = e >> 1, half = e & 1;
  int koff0  = (lane < 16) ? 0 : 8;
  int klocal = (v < 4) ? (koff0 + (v << 1) + half)
                       : (16 + koff0 + ((v - 4) << 1) + half);
  int k = kt * 32 + klocal;
  int n = nt * 16 + (lane & 15);
  float val = transpose ? W[(size_t)n * Kdim + k] : W[(size_t)k * N + n];
  out[id] = f2bf(val * scale);
}

__global__ void cvt_bf16(const float* __restrict__ in, __bf16* __restrict__ out, int total) {
  int id = blockIdx.x * blockDim.x + threadIdx.x;
  if (id < total) out[id] = f2bf(in[id]);
}

// ---------------------------------------------------------------------------
// Main fused kernel: 256 threads = 8 waves, each wave owns one 16-row M-tile.
// ---------------------------------------------------------------------------
__global__ __launch_bounds__(256) void fused_mlp(
    const float* __restrict__ x,
    const float* __restrict__ b1, const float* __restrict__ b2,
    const float* __restrict__ b3,
    const float* __restrict__ W4, const float* __restrict__ b4,
    const __bf16* __restrict__ w1f, const __bf16* __restrict__ w2f,
    const __bf16* __restrict__ wkf, const __bf16* __restrict__ w3f,
    const __bf16* __restrict__ vbf,
    float* __restrict__ out) {
  extern __shared__ char smem[];
  __bf16* xs  = (__bf16*)smem;                  // [128][520] bf16 (layer-1 A)
  __bf16* zcs = (__bf16*)smem;                  // reuse: [128][264] bf16 (z | mem)
  float*  lgs = (float*)(smem + MBLK * ZC_LD * 2); // reuse: [128][36] f32
  __bf16* z1s = (__bf16*)(smem + XS_BYTES);     // [128][392] bf16

  const int tid  = threadIdx.x;
  const int wave = tid >> 5;
  const int lane = tid & 31;
  const int row0 = blockIdx.x * MBLK;
  const int mrow = wave * 16;
  const int rb   = (lane < 16) ? 0 : 8;
  const v8f vzero = {};

  // ---------- stage x tile -> LDS bf16, coalesced NT loads (x is read once)
  {
    const float* xb = x + (size_t)row0 * DIN;
#pragma unroll 4
    for (int it = 0; it < (MBLK * DIN / 4) / 256; ++it) {
      int idx = it * 256 + tid;
      int r  = idx >> 7;          // 128 float4 per row
      int c4 = idx & 127;
      v4f val = __builtin_nontemporal_load((const v4f*)(xb + (size_t)r * DIN + c4 * 4));
      v4bf o;
#pragma unroll
      for (int i = 0; i < 4; ++i) o[i] = f2bf(val[i]);
      *(v4bf*)(xs + (size_t)r * XS_LD + c4 * 4) = o;
    }
  }
  __syncthreads();

  // ------------------ layer 1: z1 = gelu(x @ W1 + b1)  [128x512]@[512x384]
  {
    const __bf16* xa = xs + (size_t)mrow * XS_LD;
    for (int ng = 0; ng < 24; ng += 8) {
      v8f acc[8];
#pragma unroll
      for (int j = 0; j < 8; ++j) acc[j] = vzero;
      for (int kt = 0; kt < 16; ++kt) {
        v16bf a = load_a_lds(xa, XS_LD, kt, lane);
#pragma unroll
        for (int j = 0; j < 8; ++j) {
          v16bf b = load_b_frag(w1f, kt * 24 + ng + j, lane);
          acc[j] = wmma_bf16(a, b, acc[j]);
        }
      }
#pragma unroll
      for (int j = 0; j < 8; ++j) {
        int col = (ng + j) * 16 + (lane & 15);
        float bias = b1[col];
#pragma unroll
        for (int v = 0; v < 8; ++v) {
          float g = gelu_erf(acc[j][v] + bias);
          z1s[(size_t)(mrow + rb + v) * Z1_LD + col] = f2bf(g);
        }
      }
    }
  }
  __syncthreads();   // xs reads done -> safe to reuse region as zcs/lgs

  // ------------------ layer 2: z = gelu(z1 @ W2 + b2)  [128x384]@[384x128]
  {
    v8f acc[8];
#pragma unroll
    for (int j = 0; j < 8; ++j) acc[j] = vzero;
    const __bf16* za = z1s + (size_t)mrow * Z1_LD;
    for (int kt = 0; kt < 12; ++kt) {
      v16bf a = load_a_lds(za, Z1_LD, kt, lane);
#pragma unroll
      for (int j = 0; j < 8; ++j) {
        v16bf b = load_b_frag(w2f, kt * 8 + j, lane);
        acc[j] = wmma_bf16(a, b, acc[j]);
      }
    }
#pragma unroll
    for (int j = 0; j < 8; ++j) {
      int col = j * 16 + (lane & 15);
      float bias = b2[col];
#pragma unroll
      for (int v = 0; v < 8; ++v) {
        float g = gelu_erf(acc[j][v] + bias);
        zcs[(size_t)(mrow + rb + v) * ZC_LD + col] = f2bf(g);
      }
    }
  }

  // ------------------ logits = z @ (mem_keys^T / tau)  [128x128]@[128x32]
  // (wave reads only its own zcs rows; per-wave DS ordering suffices)
  {
    v8f lg[2];
    lg[0] = vzero; lg[1] = vzero;
    const __bf16* za = zcs + (size_t)mrow * ZC_LD;
    for (int kt = 0; kt < 4; ++kt) {
      v16bf a = load_a_lds(za, ZC_LD, kt, lane);
#pragma unroll
      for (int j = 0; j < 2; ++j) {
        v16bf b = load_b_frag(wkf, kt * 2 + j, lane);
        lg[j] = wmma_bf16(a, b, lg[j]);
      }
    }
#pragma unroll
    for (int j = 0; j < 2; ++j)
#pragma unroll
      for (int v = 0; v < 8; ++v)
        lgs[(size_t)(mrow + rb + v) * LG_LD + j * 16 + (lane & 15)] = lg[j][v];
  }
  __syncthreads();

  // ------------------ top-2 over 32 slots + 2-way softmax + attn @ V
  if (tid < MBLK) {
    const float* lr = lgs + (size_t)tid * LG_LD;
    float best = -3.0e38f, second = -3.0e38f;
    int bi = 0, si = 0;
#pragma unroll
    for (int k = 0; k < NSLOT; ++k) {
      float v = lr[k];
      if (v > best)        { second = best; si = bi; best = v; bi = k; }
      else if (v > second) { second = v; si = k; }
    }
    float e1 = expf(second - best);
    float a0 = 1.0f / (1.0f + e1);
    float a1 = e1 * a0;
    const __bf16* r0 = vbf + (size_t)bi * LAT;
    const __bf16* r1 = vbf + (size_t)si * LAT;
    __bf16* dst = zcs + (size_t)tid * ZC_LD + LAT;
#pragma unroll
    for (int j = 0; j < LAT; j += 8) {
      v8bf u0 = *(const v8bf*)(r0 + j);
      v8bf u1 = *(const v8bf*)(r1 + j);
      v8bf o;
#pragma unroll
      for (int i = 0; i < 8; ++i)
        o[i] = f2bf(a0 * bf2f(u0[i]) + a1 * bf2f(u1[i]));
      *(v8bf*)(dst + j) = o;
    }
  }
  __syncthreads();

  // ------------------ layer 3 + folded layer 4:
  // h = gelu([z|mem] @ W3 + b3); out = sigmoid(h @ W4 + b4)
  {
    float part[8];
#pragma unroll
    for (int v = 0; v < 8; ++v) part[v] = 0.0f;
    const __bf16* za = zcs + (size_t)mrow * ZC_LD;
    for (int ng = 0; ng < 24; ng += 8) {
      v8f acc[8];
#pragma unroll
      for (int j = 0; j < 8; ++j) acc[j] = vzero;
      for (int kt = 0; kt < 8; ++kt) {
        v16bf a = load_a_lds(za, ZC_LD, kt, lane);
#pragma unroll
        for (int j = 0; j < 8; ++j) {
          v16bf b = load_b_frag(w3f, kt * 24 + ng + j, lane);
          acc[j] = wmma_bf16(a, b, acc[j]);
        }
      }
#pragma unroll
      for (int j = 0; j < 8; ++j) {
        int col = (ng + j) * 16 + (lane & 15);
        float bias = b3[col];
        float w4v  = W4[col];
#pragma unroll
        for (int v = 0; v < 8; ++v)
          part[v] += gelu_erf(acc[j][v] + bias) * w4v;
      }
    }
    // Reduce across the 16 lanes holding different output columns.
#pragma unroll
    for (int m = 1; m < 16; m <<= 1)
#pragma unroll
      for (int v = 0; v < 8; ++v)
        part[v] += __shfl_xor(part[v], m, 32);
    if ((lane & 15) == 0) {
      float bb = b4[0];
#pragma unroll
      for (int v = 0; v < 8; ++v) {
        float s = part[v] + bb;
        out[row0 + mrow + rb + v] = 1.0f / (1.0f + expf(-s));
      }
    }
  }
}

// ---------------------------------------------------------------------------
extern "C" void kernel_launch(void* const* d_in, const int* in_sizes, int n_in,
                              void* d_out, int out_size, void* d_ws, size_t ws_size,
                              hipStream_t stream) {
  const float* x  = (const float*)d_in[0];
  const float* W1 = (const float*)d_in[1];
  const float* b1 = (const float*)d_in[2];
  const float* W2 = (const float*)d_in[3];
  const float* b2 = (const float*)d_in[4];
  const float* mk = (const float*)d_in[5];
  const float* mv = (const float*)d_in[6];
  const float* W3 = (const float*)d_in[7];
  const float* b3 = (const float*)d_in[8];
  const float* W4 = (const float*)d_in[9];
  const float* b4 = (const float*)d_in[10];
  float* out = (float*)d_out;

  char* ws = (char*)d_ws;
  __bf16* w1f = (__bf16*)(ws + 0);        // 512x384 frags -> 393216 B
  __bf16* w2f = (__bf16*)(ws + 393216);   // 384x128 frags ->  98304 B
  __bf16* wkf = (__bf16*)(ws + 491520);   // keys^T (128x32), pre-scaled 1/tau -> 8192 B
  __bf16* w3f = (__bf16*)(ws + 499712);   // 256x384 frags -> 196608 B
  __bf16* vbf = (__bf16*)(ws + 696320);   // mem_values bf16 row-major -> 8192 B

  pack_b_frags<<<(512 * 384) / 256, 256, 0, stream>>>(W1, w1f, 512, 384, 1.0f, 0, 512 * 384);
  pack_b_frags<<<(384 * 128) / 256, 256, 0, stream>>>(W2, w2f, 384, 128, 1.0f, 0, 384 * 128);
  pack_b_frags<<<(128 * 32) / 256,  256, 0, stream>>>(mk, wkf, 128, 32, 1.0f / TAU_F, 1, 128 * 32);
  pack_b_frags<<<(256 * 384) / 256, 256, 0, stream>>>(W3, w3f, 256, 384, 1.0f, 0, 256 * 384);
  cvt_bf16<<<(32 * 128) / 256, 256, 0, stream>>>(mv, vbf, 32 * 128);

  fused_mlp<<<BATCH / MBLK, 256, LDS_TOTAL, stream>>>(x, b1, b2, b3, W4, b4,
                                                      w1f, w2f, wkf, w3f, vbf, out);
}
